// SelfAttentionHead_65146063946135
// MI455X (gfx1250) — matrix operand order repaired and checked by reading, not compiled, not run on hardware
//
#include <hip/hip_runtime.h>

#define S_LEN 2048
#define D_DIM 1024
#define B_NUM 4

typedef __attribute__((ext_vector_type(16))) __bf16 v16bf;
typedef __attribute__((ext_vector_type(8)))  float  v8f;

union FragBF {
  v16bf v;
  unsigned int u[8];
};

#define WMMA_BF16(Afrag, Bfrag, Cacc) \
  __builtin_amdgcn_wmma_f32_16x16x32_bf16(false, (Afrag), false, (Bfrag), (short)0, (Cacc), false, false)

static __device__ __forceinline__ unsigned int bf16_bits(float f) {
  unsigned int u = __float_as_uint(f);
  return (u + 0x7FFFu + ((u >> 16) & 1u)) >> 16;  // round-to-nearest-even
}
static __device__ __forceinline__ unsigned int pack2bf(float lo, float hi) {
  return (bf16_bits(lo) & 0xFFFFu) | (bf16_bits(hi) << 16);
}

// out[m][n] = X[m][:] @ W[:][n] + bias[n]; M = B*S = 8192, N = K = 1024.
// Writes optional f32 copy and a bf16 copy (optionally transposed to [b][d][s]).
__global__ __launch_bounds__(256) void proj_gemm_bf16(
    const float* __restrict__ X, const float* __restrict__ W,
    const float* __restrict__ bias,
    float* __restrict__ out_f32,
    unsigned short* __restrict__ out_b16,
    int transpose_out)
{
  __shared__ unsigned int Xs[128 * 16];  // 128 rows x 32 bf16 (16 packed dwords)
  __shared__ unsigned int Ws[64 * 16];   // 64 cols  x 32 bf16, stored [n][k]

  const int tid  = threadIdx.x;
  const int lane = tid & 31;
  const int wv   = tid >> 5;       // wave 0..7 -> 16-row strip
  const int half = lane >> 4;      // K-half selector per WMMA layout
  const int lm   = lane & 15;
  const int row0 = blockIdx.y * 128;
  const int n0b  = blockIdx.x * 64;

  const v8f vzero = {0.f,0.f,0.f,0.f,0.f,0.f,0.f,0.f};
  v8f acc[4];
  acc[0] = vzero; acc[1] = vzero; acc[2] = vzero; acc[3] = vzero;

  const int wn = tid & 63;   // W-tile loader: column within tile
  const int wk = tid >> 6;   // 0..3 -> k-subrange of 8

  for (int f0 = 0; f0 < D_DIM; f0 += 32) {
    // Stage X tile 128x32 f32 -> bf16 LDS (float4 coalesced loads)
    #pragma unroll
    for (int i = 0; i < 4; ++i) {
      int idx = tid + i * 256;
      int r   = idx >> 3;
      int c4  = (idx & 7) << 2;
      const float4 xv = *reinterpret_cast<const float4*>(
          X + (size_t)(row0 + r) * D_DIM + f0 + c4);
      Xs[r * 16 + (c4 >> 1) + 0] = pack2bf(xv.x, xv.y);
      Xs[r * 16 + (c4 >> 1) + 1] = pack2bf(xv.z, xv.w);
    }
    // Stage W tile 32x64 transposed -> Ws[n][k] (packed along k)
    #pragma unroll
    for (int i = 0; i < 8; i += 2) {
      float wlo = W[(size_t)(f0 + wk * 8 + i)     * D_DIM + n0b + wn];
      float whi = W[(size_t)(f0 + wk * 8 + i + 1) * D_DIM + n0b + wn];
      Ws[wn * 16 + wk * 4 + (i >> 1)] = pack2bf(wlo, whi);
    }
    // Prefetch next K-step of W rows into the cache hierarchy
    if (f0 + 32 < D_DIM)
      __builtin_prefetch(W + (size_t)(f0 + 32 + wk * 8) * D_DIM + n0b + wn, 0, 0);
    __syncthreads();

    FragBF a;
    #pragma unroll
    for (int j = 0; j < 8; ++j) {
      int jj = ((j < 4) ? j : j + 4) + (half ? 4 : 0);
      a.u[j] = Xs[(wv * 16 + lm) * 16 + jj];
    }
    #pragma unroll
    for (int t = 0; t < 4; ++t) {
      FragBF bm;
      #pragma unroll
      for (int j = 0; j < 8; ++j)
        bm.u[j] = Ws[(t * 16 + lm) * 16 + ((j < 4) ? j : j + 4) + (half ? 4 : 0)];
      acc[t] = WMMA_BF16(a.v, bm.v, acc[t]);
    }
    __syncthreads();
  }

  #pragma unroll
  for (int t = 0; t < 4; ++t) {
    const int col  = n0b + t * 16 + lm;
    const float bval = bias[col];
    #pragma unroll
    for (int e = 0; e < 8; ++e) {
      const int grow = row0 + wv * 16 + e + half * 8;
      const float val = acc[t][e] + bval;
      if (out_f32) out_f32[(size_t)grow * D_DIM + col] = val;
      if (transpose_out) {
        const int bb = grow >> 11;           // grow / S_LEN
        const int ss = grow & (S_LEN - 1);
        out_b16[((size_t)bb * D_DIM + col) * S_LEN + ss] = (unsigned short)bf16_bits(val);
      } else {
        out_b16[(size_t)grow * D_DIM + col] = (unsigned short)bf16_bits(val);
      }
    }
  }
}

// One block per (batch, 16-row q tile). Exact causal softmax, WMMA QK^T and PV.
__global__ __launch_bounds__(256) void attn_fused(
    const unsigned int* __restrict__ qp_b16,   // [B,S,D] bf16 (dword view)
    const unsigned int* __restrict__ kp_b16,   // [B,S,D] bf16
    const unsigned int* __restrict__ vpT_b16,  // [B,D,S] bf16 (transposed)
    const float* __restrict__ qp_f32,          // [B,S,D] f32 residual
    float* __restrict__ out)                   // [B,S,D] f32
{
  extern __shared__ unsigned char smem_raw[];
  unsigned int* Qs  = reinterpret_cast<unsigned int*>(smem_raw);                 // 16 x 512 dwords (32 KB)
  float*        Sb  = reinterpret_cast<float*>(smem_raw + 16 * 512 * 4);         // 16 x 2048 f32 (128 KB)
  float*        lrow= reinterpret_cast<float*>(smem_raw + 16 * 512 * 4 + 16 * 2048 * 4);

  const int tid  = threadIdx.x;
  const int lane = tid & 31;
  const int wv   = tid >> 5;
  const int half = lane >> 4;
  const int lm   = lane & 15;

  const int bb = blockIdx.x >> 7;       // batch
  const int qt = blockIdx.x & 127;      // q tile index
  const int q0 = qt * 16;
  const int ncols   = (qt + 1) * 16;    // causal key count
  const int ncols32 = (ncols + 31) & ~31;

  // Phase A: Q tile -> LDS via async direct global->LDS DMA (ASYNCcnt path).
  // 2048 16-byte chunks, 8 per thread; both LDS and global addresses 16B aligned.
  {
    const size_t base = ((size_t)bb * S_LEN + q0) * (D_DIM / 2);
    #pragma unroll
    for (int i = 0; i < 8; ++i) {
      const int idx4 = tid + i * 256;
      // Low 32 bits of a generic pointer into LDS == LDS-relative byte offset.
      unsigned lds_addr = (unsigned)(size_t)(Qs + (size_t)idx4 * 4);
      const unsigned int* gsrc = qp_b16 + base + (size_t)idx4 * 4;
      asm volatile("global_load_async_to_lds_b128 %0, %1, off"
                   :: "v"(lds_addr), "v"(gsrc) : "memory");
    }
    asm volatile("s_wait_asynccnt 0" ::: "memory");
  }
  __syncthreads();

  const v8f vzero = {0.f,0.f,0.f,0.f,0.f,0.f,0.f,0.f};

  // Phase B: scores S = (Q K^T) * 1/sqrt(D), causal mask, store f32 to LDS
  for (int kt = wv; kt <= qt; kt += 8) {
    const int k0 = kt * 16;
    v8f acc = vzero;
    const size_t kbase = ((size_t)bb * S_LEN + k0 + lm) * (D_DIM / 2);
    // Prefetch this wave's next key-tile row (L2-resident -> WGP$)
    if (kt + 8 <= qt)
      __builtin_prefetch(kp_b16 + kbase + (size_t)128 * (D_DIM / 2), 0, 0);
    for (int f0 = 0; f0 < D_DIM; f0 += 32) {
      FragBF a, bm;
      const int f2 = f0 >> 1;
      #pragma unroll
      for (int j = 0; j < 8; ++j) {
        const int jj = ((j < 4) ? j : j + 4) + (half ? 4 : 0);
        a.u[j]  = Qs[lm * (D_DIM / 2) + f2 + jj];
        bm.u[j] = kp_b16[kbase + f2 + jj];
      }
      acc = WMMA_BF16(a.v, bm.v, acc);
    }
    #pragma unroll
    for (int e = 0; e < 8; ++e) {
      const int row  = e + half * 8;
      const int gcol = k0 + lm;
      float sc = acc[e] * 0.03125f;          // 1/sqrt(1024)
      if (gcol > q0 + row) sc = -1.0e10f;    // causal mask (matches reference)
      Sb[row * S_LEN + gcol] = sc;
    }
  }
  __syncthreads();

  // Phase C: softmax numerator per row (keep 1/l for epilogue); pad to 32 keys
  #pragma unroll
  for (int rr = 0; rr < 2; ++rr) {
    const int row = wv * 2 + rr;
    float mx = -3.0e38f;
    for (int c = lane; c < ncols; c += 32)
      mx = fmaxf(mx, Sb[row * S_LEN + c]);
    #pragma unroll
    for (int off = 16; off > 0; off >>= 1)
      mx = fmaxf(mx, __shfl_xor(mx, off, 32));
    float sum = 0.f;
    for (int c = lane; c < ncols; c += 32) {
      const float p = __expf(Sb[row * S_LEN + c] - mx);
      Sb[row * S_LEN + c] = p;
      sum += p;
    }
    #pragma unroll
    for (int off = 16; off > 0; off >>= 1)
      sum += __shfl_xor(sum, off, 32);
    if (lane == 0) lrow[row] = sum;
    // zero-pad up to a 32-key multiple for the PV WMMA loop
    for (int c = ncols + lane; c < ncols32; c += 32)
      Sb[row * S_LEN + c] = 0.f;
  }
  __syncthreads();

  // Phase D: O = P @ V; wave wv owns output dims [wv*128, wv*128+128)
  const int n0 = wv * 128;
  v8f acc[8];
  #pragma unroll
  for (int t = 0; t < 8; ++t) acc[t] = vzero;

  for (int kk = 0; kk < ncols32; kk += 32) {
    // Prefetch next key-block of V for this wave's dim slice
    if (kk + 32 < ncols32)
      __builtin_prefetch(
          vpT_b16 + ((((size_t)bb * D_DIM + n0 + lm) * S_LEN + kk + 32) >> 1), 0, 0);
    FragBF a;
    #pragma unroll
    for (int j = 0; j < 8; ++j) {
      const int jj   = ((j < 4) ? j : j + 4) + (half ? 4 : 0);
      const int kidx = kk + 2 * jj;
      a.u[j] = pack2bf(Sb[lm * S_LEN + kidx], Sb[lm * S_LEN + kidx + 1]);
    }
    #pragma unroll
    for (int t = 0; t < 8; ++t) {
      const int n = n0 + t * 16 + lm;
      const size_t vb = (((size_t)bb * D_DIM + n) * S_LEN + kk) >> 1;
      FragBF bm;
      #pragma unroll
      for (int j = 0; j < 8; ++j)
        bm.u[j] = vpT_b16[vb + ((j < 4) ? j : j + 4) + (half ? 4 : 0)];
      acc[t] = WMMA_BF16(a.v, bm.v, acc[t]);
    }
  }

  // Epilogue: divide by row sum, add qp residual (f32)
  #pragma unroll
  for (int t = 0; t < 8; ++t) {
    const int col = n0 + t * 16 + lm;
    #pragma unroll
    for (int e = 0; e < 8; ++e) {
      const int row = e + half * 8;
      const float inv = 1.0f / lrow[row];
      const size_t oidx = ((size_t)bb * S_LEN + q0 + row) * D_DIM + col;
      out[oidx] = qp_f32[oidx] + acc[t][e] * inv;
    }
  }
}

extern "C" void kernel_launch(void* const* d_in, const int* in_sizes, int n_in,
                              void* d_out, int out_size, void* d_ws, size_t ws_size,
                              hipStream_t stream) {
  (void)in_sizes; (void)n_in; (void)out_size; (void)ws_size;
  const float* v  = (const float*)d_in[0];
  const float* k  = (const float*)d_in[1];
  const float* q  = (const float*)d_in[2];
  // d_in[3] = mask (causal tril; enforced analytically in-kernel)
  const float* Wq = (const float*)d_in[4];
  const float* bq = (const float*)d_in[5];
  const float* Wk = (const float*)d_in[6];
  const float* bk = (const float*)d_in[7];
  const float* Wv = (const float*)d_in[8];
  const float* bv = (const float*)d_in[9];
  float* out = (float*)d_out;

  // Workspace layout (80 MB total):
  //   [0,32MB)   qp f32   [32,48MB) qp bf16   [48,64MB) kp bf16   [64,80MB) vp^T bf16
  char* ws = (char*)d_ws;
  float*          qp_f32  = (float*)ws;
  unsigned short* qp_b16  = (unsigned short*)(ws + (size_t)33554432);
  unsigned short* kp_b16  = (unsigned short*)(ws + (size_t)33554432 + (size_t)16777216);
  unsigned short* vpT_b16 = (unsigned short*)(ws + (size_t)33554432 + (size_t)2 * 16777216);

  dim3 blk(256, 1, 1);
  dim3 pgrid(D_DIM / 64, (B_NUM * S_LEN) / 128, 1);  // (16, 64)
  proj_gemm_bf16<<<pgrid, blk, 0, stream>>>(q, Wq, bq, qp_f32, qp_b16, 0);
  proj_gemm_bf16<<<pgrid, blk, 0, stream>>>(k, Wk, bk, nullptr, kp_b16, 0);
  proj_gemm_bf16<<<pgrid, blk, 0, stream>>>(v, Wv, bv, nullptr, vpT_b16, 1);

  const size_t smem = 16 * 512 * 4 + 16 * 2048 * 4 + 16 * 4;  // 163,904 B dynamic LDS
  attn_fused<<<dim3(B_NUM * (S_LEN / 16), 1, 1), blk, smem, stream>>>(
      (const unsigned int*)qp_b16, (const unsigned int*)kp_b16,
      (const unsigned int*)vpT_b16, qp_f32, out);
}
